// GeneralNonLinearReadoutBlock_6605659702015
// MI455X (gfx1250) — compile-verified, matrix operand order checked by reading, not run on hardware
//
#include <hip/hip_runtime.h>
#include <hip/hip_bf16.h>

// ---------------------------------------------------------------------------
// GeneralNonLinearReadoutBlock, fused, bf16-WMMA (gfx1250 / CDNA5, wave32)
//   x:(N,512) f32 ; w1_s:(128,256) w1_v,w2_s,w2_v:(128,128) f32 ; out:(N,512) f32
// ---------------------------------------------------------------------------

#define MULDIM 128
#define ROWLEN 512                       // 128 + 3*128
#define HSTR   136                       // padded bf16 row stride (136*2B = 68 words % 64 = 4)
#define GSTR   130                       // padded f32  row stride (8*130 % 64 = 16)
static __device__ __host__ constexpr float INV_SQRT_FANIN = 0.08838834764831845f;

typedef __attribute__((ext_vector_type(16))) __bf16 v16bf;
typedef __attribute__((ext_vector_type(8)))  __bf16 v8bf;
typedef __attribute__((ext_vector_type(8)))  float  v8f;
typedef __attribute__((ext_vector_type(4)))  float  v4f;

union BF16x16 { v16bf v; unsigned u[8]; };

// packed round-to-nearest-even: two f32 -> one dword holding (bf16(lo) | bf16(hi)<<16)
__device__ __forceinline__ unsigned pack2bf(float lo, float hi) {
    unsigned ul = __builtin_bit_cast(unsigned, lo);
    unsigned uh = __builtin_bit_cast(unsigned, hi);
    ul = (ul + 0x7FFFu + ((ul >> 16) & 1u)) >> 16;
    uh = (uh + 0x7FFFu + ((uh >> 16) & 1u)) & 0xFFFF0000u;
    return ul | uh;
}

__device__ __forceinline__ __bf16 f2bf(float f) {
    unsigned u = __builtin_bit_cast(unsigned, f);
    unsigned short s = (unsigned short)((u + 0x7FFFu + ((u >> 16) & 1u)) >> 16);
    return __builtin_bit_cast(__bf16, s);
}

__device__ __forceinline__ v8f wmma_bf16(v16bf a, v16bf b, v8f c) {
    return __builtin_amdgcn_wmma_f32_16x16x32_bf16(
        false, a, false, b, (short)0, c, false, false);
}

// fast SiLU: x * rcp(1 + exp(-x))   (v_exp_f32 + v_rcp_f32)
__device__ __forceinline__ float silu(float f) {
    float e = __expf(-f);
    return f * __builtin_amdgcn_rcpf(1.0f + e);
}

// A fragment (16x32 bf16) from an f32 row; kbase = kc*32 + 8*(lane>=16)
__device__ __forceinline__ v16bf load_a_frag_f32(const float* __restrict__ rowp, int kbase) {
    const v4f* p = reinterpret_cast<const v4f*>(rowp + kbase);
    const v4f* q = reinterpret_cast<const v4f*>(rowp + kbase + 16);
    v4f A = p[0], B = p[1], C = q[0], D = q[1];
    BF16x16 f;
    f.u[0] = pack2bf(A[0], A[1]);  f.u[1] = pack2bf(A[2], A[3]);
    f.u[2] = pack2bf(B[0], B[1]);  f.u[3] = pack2bf(B[2], B[3]);
    f.u[4] = pack2bf(C[0], C[1]);  f.u[5] = pack2bf(C[2], C[3]);
    f.u[6] = pack2bf(D[0], D[1]);  f.u[7] = pack2bf(D[2], D[3]);
    return f.v;
}

// B fragment (Kx16) == A pattern applied to B^T rows (wT: N x 128, row-major);
// also used to reload staged activations (row stride `stride` bf16) as A frags.
__device__ __forceinline__ v16bf load_bf_frag(const __bf16* row, int kbase) {
    v8bf lo = *reinterpret_cast<const v8bf*>(row + kbase);
    v8bf hi = *reinterpret_cast<const v8bf*>(row + kbase + 16);
    return __builtin_shufflevector(lo, hi, 0, 1, 2, 3, 4, 5, 6, 7,
                                           8, 9, 10, 11, 12, 13, 14, 15);
}

// A fragment for x1 component i (stride-3 floats); rowp = x + row*512 + 128 + i
__device__ __forceinline__ v16bf load_x1_frag(const float* __restrict__ rowp, int kbase) {
    BF16x16 f;
#pragma unroll
    for (int j = 0; j < 4; ++j)
        f.u[j] = pack2bf(rowp[(kbase + 2 * j) * 3], rowp[(kbase + 2 * j + 1) * 3]);
#pragma unroll
    for (int j = 0; j < 4; ++j)
        f.u[4 + j] = pack2bf(rowp[(kbase + 16 + 2 * j) * 3], rowp[(kbase + 17 + 2 * j) * 3]);
    return f.v;
}

// ---------------------------------------------------------------------------
// Prep: 4 weight matrices -> bf16, transposed (N-major, K=128 contiguous)
// ws layout:  w1sT[256*128] | w1vT[128*128] | w2sT[128*128] | w2vT[128*128]
// ---------------------------------------------------------------------------
__global__ void prep_weights(const float* __restrict__ w1s, const float* __restrict__ w1v,
                             const float* __restrict__ w2s, const float* __restrict__ w2v,
                             unsigned short* __restrict__ wT_) {
    __bf16* wT = reinterpret_cast<__bf16*>(wT_);
    int idx = blockIdx.x * blockDim.x + threadIdx.x;   // 0 .. 81919
    if (idx < 256 * 128) {
        int n = idx >> 7, k = idx & 127;
        wT[idx] = f2bf(w1s[k * 256 + n]);
    } else if (idx < 256 * 128 + 128 * 128) {
        int j = idx - 256 * 128; int n = j >> 7, k = j & 127;
        wT[idx] = f2bf(w1v[k * 128 + n]);
    } else if (idx < 256 * 128 + 2 * 128 * 128) {
        int j = idx - (256 * 128 + 128 * 128); int n = j >> 7, k = j & 127;
        wT[idx] = f2bf(w2s[k * 128 + n]);
    } else if (idx < 256 * 128 + 3 * 128 * 128) {
        int j = idx - (256 * 128 + 2 * 128 * 128); int n = j >> 7, k = j & 127;
        wT[idx] = f2bf(w2v[k * 128 + n]);
    }
}

// ---------------------------------------------------------------------------
// Fused main kernel: 8 waves / block, one 16-row tile per wave per iteration.
// ---------------------------------------------------------------------------
__global__ __launch_bounds__(256) void fused_readout(
    const float* __restrict__ x, const unsigned short* __restrict__ wT_,
    float* __restrict__ out, int N, int numTiles) {

    __shared__ __bf16 hstage[8][16 * HSTR];   // 4.25 KB/wave: D-layout -> A-layout transpose
    __shared__ float  gstage[8][16 * GSTR];   // 8.1  KB/wave: gate values (f32)

    const __bf16* wT   = reinterpret_cast<const __bf16*>(wT_);
    const __bf16* w1sT = wT;
    const __bf16* w1vT = wT + 256 * 128;
    const __bf16* w2sT = w1vT + 128 * 128;
    const __bf16* w2vT = w2sT + 128 * 128;

    const int lane  = threadIdx.x & 31;
    const int wave  = threadIdx.x >> 5;
    const int halfw = lane >> 4;          // 0: lanes 0-15, 1: lanes 16-31
    const int mrow  = lane & 15;          // A-row index / D-column index for this lane
    const int kb    = halfw * 8;          // per-lane K sub-offset inside a 32-chunk
    __bf16* myh = &hstage[wave][0];
    float*  myg = &gstage[wave][0];
    const int dbase = 8 * halfw;          // D-layout row offset for this lane half

    for (int t = blockIdx.x * 8 + wave; t < numTiles; t += gridDim.x * 8) {
        const int base = t * 16;
        int arow = base + mrow; if (arow >= N) arow = N - 1;     // clamp tail loads
        const float* xrow = x + (size_t)arow * ROWLEN;

        // ---- load x0 A fragments (K = 128 -> 4 chunks)
        v16bf a[4];
#pragma unroll
        for (int kc = 0; kc < 4; ++kc) a[kc] = load_a_frag_f32(xrow, kc * 32 + kb);

        // ---- stage 1a: h0 tiles (s cols 0..127) -> SiLU -> LDS (bf16, A-reloadable)
#pragma unroll 2
        for (int nt = 0; nt < 8; ++nt) {
            const __bf16* wrow = w1sT + (nt * 16 + mrow) * 128;
            v16bf b[4];
#pragma unroll
            for (int kc = 0; kc < 4; ++kc) b[kc] = load_bf_frag(wrow, kc * 32 + kb);
            v8f acc = {0.f, 0.f, 0.f, 0.f, 0.f, 0.f, 0.f, 0.f};
#pragma unroll
            for (int kc = 0; kc < 4; ++kc) acc = wmma_bf16(a[kc], b[kc], acc);
#pragma unroll
            for (int r = 0; r < 8; ++r)
                myh[(dbase + r) * HSTR + nt * 16 + mrow] =
                    f2bf(silu(acc[r] * INV_SQRT_FANIN));
        }

        // ---- stage 1b: g tiles (s cols 128..255) -> SiLU -> LDS (f32)
#pragma unroll 2
        for (int nt = 8; nt < 16; ++nt) {
            const __bf16* wrow = w1sT + (nt * 16 + mrow) * 128;
            v16bf b[4];
#pragma unroll
            for (int kc = 0; kc < 4; ++kc) b[kc] = load_bf_frag(wrow, kc * 32 + kb);
            v8f acc = {0.f, 0.f, 0.f, 0.f, 0.f, 0.f, 0.f, 0.f};
#pragma unroll
            for (int kc = 0; kc < 4; ++kc) acc = wmma_bf16(a[kc], b[kc], acc);
#pragma unroll
            for (int r = 0; r < 8; ++r)
                myg[(dbase + r) * GSTR + (nt - 8) * 16 + mrow] =
                    silu(acc[r] * INV_SQRT_FANIN);
        }

        // ---- stage 2: y0 = h0 @ w2_s
        v16bf ah[4];
#pragma unroll
        for (int kc = 0; kc < 4; ++kc) ah[kc] = load_bf_frag(myh + mrow * HSTR, kc * 32 + kb);
#pragma unroll 2
        for (int nt = 0; nt < 8; ++nt) {
            const __bf16* wrow = w2sT + (nt * 16 + mrow) * 128;
            v16bf b[4];
#pragma unroll
            for (int kc = 0; kc < 4; ++kc) b[kc] = load_bf_frag(wrow, kc * 32 + kb);
            v8f acc = {0.f, 0.f, 0.f, 0.f, 0.f, 0.f, 0.f, 0.f};
#pragma unroll
            for (int kc = 0; kc < 4; ++kc) acc = wmma_bf16(ah[kc], b[kc], acc);
#pragma unroll
            for (int r = 0; r < 8; ++r) {
                int row = base + dbase + r;
                if (row < N)
                    out[(size_t)row * ROWLEN + nt * 16 + mrow] = acc[r] * INV_SQRT_FANIN;
            }
        }

        // ---- stage 3: per component i: v_i = x1_i @ w1_v ; h1_i = v_i * g ;
        //               y1_i = h1_i @ w2_v
#pragma unroll 1
        for (int i = 0; i < 3; ++i) {
            const float* x1p = xrow + MULDIM + i;
#pragma unroll
            for (int kc = 0; kc < 4; ++kc) a[kc] = load_x1_frag(x1p, kc * 32 + kb);

#pragma unroll 2
            for (int nt = 0; nt < 8; ++nt) {
                const __bf16* wrow = w1vT + (nt * 16 + mrow) * 128;
                v16bf b[4];
#pragma unroll
                for (int kc = 0; kc < 4; ++kc) b[kc] = load_bf_frag(wrow, kc * 32 + kb);
                v8f acc = {0.f, 0.f, 0.f, 0.f, 0.f, 0.f, 0.f, 0.f};
#pragma unroll
                for (int kc = 0; kc < 4; ++kc) acc = wmma_bf16(a[kc], b[kc], acc);
#pragma unroll
                for (int r = 0; r < 8; ++r) {
                    float gv = myg[(dbase + r) * GSTR + nt * 16 + mrow];
                    myh[(dbase + r) * HSTR + nt * 16 + mrow] =
                        f2bf(acc[r] * INV_SQRT_FANIN * gv);
                }
            }

#pragma unroll
            for (int kc = 0; kc < 4; ++kc) ah[kc] = load_bf_frag(myh + mrow * HSTR, kc * 32 + kb);
#pragma unroll 2
            for (int nt = 0; nt < 8; ++nt) {
                const __bf16* wrow = w2vT + (nt * 16 + mrow) * 128;
                v16bf b[4];
#pragma unroll
                for (int kc = 0; kc < 4; ++kc) b[kc] = load_bf_frag(wrow, kc * 32 + kb);
                v8f acc = {0.f, 0.f, 0.f, 0.f, 0.f, 0.f, 0.f, 0.f};
#pragma unroll
                for (int kc = 0; kc < 4; ++kc) acc = wmma_bf16(ah[kc], b[kc], acc);
#pragma unroll
                for (int r = 0; r < 8; ++r) {
                    int row = base + dbase + r;
                    if (row < N)
                        out[(size_t)row * ROWLEN + MULDIM + (nt * 16 + mrow) * 3 + i] =
                            acc[r] * INV_SQRT_FANIN;
                }
            }
        }
    }
}

// ---------------------------------------------------------------------------
extern "C" void kernel_launch(void* const* d_in, const int* in_sizes, int n_in,
                              void* d_out, int out_size, void* d_ws, size_t ws_size,
                              hipStream_t stream) {
    const float* x   = (const float*)d_in[0];
    const float* w1s = (const float*)d_in[1];
    const float* w1v = (const float*)d_in[2];
    const float* w2s = (const float*)d_in[3];
    const float* w2v = (const float*)d_in[4];
    float* out = (float*)d_out;
    unsigned short* wT = (unsigned short*)d_ws;    // 81920 bf16 = 160 KB scratch

    const int N = in_sizes[0] / ROWLEN;
    const int numTiles = (N + 15) / 16;

    prep_weights<<<320, 256, 0, stream>>>(w1s, w1v, w2s, w2v, wT);   // 81920 elems

    int blocks = (numTiles + 7) / 8;               // 8 waves (tiles) per block
    if (blocks > 65535) blocks = 65535;
    fused_readout<<<blocks, 256, 0, stream>>>(x, wT, out, N, numTiles);
}
